// SimpleLanguageModel_65498251264578
// MI455X (gfx1250) — matrix operand order repaired and checked by reading, compiled, not run on hardware
//
#include <hip/hip_runtime.h>
#include <hip/hip_bf16.h>

// ---------------------------------------------------------------------------
// Problem constants (reference: B=2, T=2048, H=1024, V=32000, L=4)
// ---------------------------------------------------------------------------
static constexpr int Bb = 2;
static constexpr int Tt = 2048;
static constexpr int Hh = 1024;
static constexpr int Vv = 32000;
static constexpr int Ll = 4;
static constexpr int Mm = Bb * Tt; // 4096 rows of activations

typedef __attribute__((ext_vector_type(16))) __bf16 v16bf;
typedef __attribute__((ext_vector_type(8)))  __bf16 v8bf;
typedef __attribute__((ext_vector_type(8)))  float  v8f;

union FragBF {
    v16bf  v;
    v8bf   h[2];
    __bf16 e[16];
};
union AccF {
    v8f   v;
    float f[8];
};

// ---------------------------------------------------------------------------
// Embedding: x[b,t,:] = tok_emb[idx[b,t],:] + pos_emb[t,:]  -> bf16
// ---------------------------------------------------------------------------
__global__ void __launch_bounds__(256)
embed_kernel(const int* __restrict__ idx, const float* __restrict__ tok,
             const float* __restrict__ pos, __bf16* __restrict__ xout) {
    const int row   = blockIdx.x;        // 0..4095 (b*T + t)
    const int tpos  = row & (Tt - 1);
    const int token = idx[row];
    const float* te = tok + (size_t)token * Hh;
    const float* pe = pos + (size_t)tpos * Hh;
    __bf16* xo = xout + (size_t)row * Hh;
    for (int h = threadIdx.x; h < Hh; h += 256)
        xo[h] = (__bf16)(te[h] + pe[h]);
}

// ---------------------------------------------------------------------------
// Transpose f32 [R][C] -> bf16 [C][R]   (weight prep: gives K-contiguous B)
// ---------------------------------------------------------------------------
__global__ void __launch_bounds__(256)
transpose_f32_to_bf16(const float* __restrict__ in, __bf16* __restrict__ out,
                      int R, int C) {
    __shared__ float s[32][33];
    const int t  = threadIdx.x;
    const int tx = t & 31, ty = t >> 5;           // 32 x 8
    const int r0 = blockIdx.y * 32, c0 = blockIdx.x * 32;
#pragma unroll
    for (int j = 0; j < 4; ++j)
        s[ty + 8 * j][tx] = in[(size_t)(r0 + ty + 8 * j) * C + c0 + tx];
    __syncthreads();
#pragma unroll
    for (int j = 0; j < 4; ++j)
        out[(size_t)(c0 + ty + 8 * j) * R + r0 + tx] = (__bf16)s[tx][ty + 8 * j];
}

// Transpose bf16 [R][C] -> bf16 [C][R], batched over blockIdx.z (V -> V^T)
__global__ void __launch_bounds__(256)
transpose_bf16(const __bf16* __restrict__ in, __bf16* __restrict__ out,
               int R, int C) {
    __shared__ __bf16 s[32][33];
    const size_t off = (size_t)blockIdx.z * R * C;
    const int t  = threadIdx.x;
    const int tx = t & 31, ty = t >> 5;
    const int r0 = blockIdx.y * 32, c0 = blockIdx.x * 32;
#pragma unroll
    for (int j = 0; j < 4; ++j)
        s[ty + 8 * j][tx] = in[off + (size_t)(r0 + ty + 8 * j) * C + c0 + tx];
    __syncthreads();
#pragma unroll
    for (int j = 0; j < 4; ++j)
        out[off + (size_t)(c0 + ty + 8 * j) * R + r0 + tx] = s[tx][ty + 8 * j];
}

// ---------------------------------------------------------------------------
// Tiled WMMA GEMM: C[M][N] = act(A[M][K] @ Bt[N][K]^T + bias)
//   bf16 inputs, f32 accumulate (v_wmma_f32_16x16x32_bf16).
//   Block tile 128x128x32, 8 waves (2x4), per-wave 64x32 = 4x2 WMMA tiles.
//   K is compile-time (always 1024 here): loop unrolls, double-buffered LDS
//   (one barrier per K-step), register prefetch of next K-panel during
//   compute, global_prefetch (gfx1250 global_prefetch_b8) two panels ahead.
//   All dims divide the tiles (M=4096, N in {1024,32000}, K=1024).
// ---------------------------------------------------------------------------
template <int K, bool OUT_BF16, bool BIAS, bool LEAKY>
__global__ void __launch_bounds__(256)
gemm_bf16_wmma(const __bf16* __restrict__ A, const __bf16* __restrict__ Bt,
               const float* __restrict__ bias, void* __restrict__ Cout,
               int M, int N) {
    __shared__ __bf16 sA[2][128][40];  // padded rows: 80B stride, 16B chunks
    __shared__ __bf16 sB[2][128][40];

    const int t     = threadIdx.x;
    const int lane  = t & 31;
    const int wid   = t >> 5;
    const int half  = lane >> 4;
    const int r16   = lane & 15;
    const int waveM = wid & 1;   // 0..1
    const int waveN = wid >> 1;  // 0..3

    const int m0 = blockIdx.y * 128;
    const int n0 = blockIdx.x * 128;

    const v8f vz = {};
    v8f acc[4][2];
#pragma unroll
    for (int m = 0; m < 4; ++m)
#pragma unroll
        for (int n = 0; n < 2; ++n) acc[m][n] = vz;

    const int ldRow = t >> 1;        // 0..127
    const int ldSeg = (t & 1) * 16;  // 0 or 16 (bf16 elements)
    const __bf16* gA = A  + (size_t)(m0 + ldRow) * K + ldSeg;
    const __bf16* gB = Bt + (size_t)(n0 + ldRow) * K + ldSeg;

    // Preload K-panel 0 into LDS buffer 0.
    uint4 ra0 = ((const uint4*)gA)[0], ra1 = ((const uint4*)gA)[1];
    uint4 rb0 = ((const uint4*)gB)[0], rb1 = ((const uint4*)gB)[1];
    {
        uint4* da = (uint4*)&sA[0][ldRow][ldSeg];
        da[0] = ra0; da[1] = ra1;
        uint4* db = (uint4*)&sB[0][ldRow][ldSeg];
        db[0] = rb0; db[1] = rb1;
    }
    __syncthreads();

    constexpr int nk = K / 32;       // 32: compile-time trip count
#pragma unroll 2
    for (int kt = 0; kt < nk; ++kt) {
        const int cur = kt & 1;
        // Issue global loads for the next panel before computing this one.
        if (kt + 1 < nk) {
            const __bf16* pa = gA + (size_t)(kt + 1) * 32;
            const __bf16* pb = gB + (size_t)(kt + 1) * 32;
            ra0 = ((const uint4*)pa)[0]; ra1 = ((const uint4*)pa)[1];
            rb0 = ((const uint4*)pb)[0]; rb1 = ((const uint4*)pb)[1];
            if (kt + 2 < nk) {
                __builtin_prefetch(gA + (size_t)(kt + 2) * 32, 0, 1);
                __builtin_prefetch(gB + (size_t)(kt + 2) * 32, 0, 1);
            }
        }

        // Compute on buffer `cur`.
        FragBF fb[2];
#pragma unroll
        for (int n = 0; n < 2; ++n) {   // B frag: lane holds col N, K=16*half+e
            const int col = waveN * 32 + n * 16 + r16;
            fb[n].h[0] = *(const v8bf*)&sB[cur][col][16 * half];
            fb[n].h[1] = *(const v8bf*)&sB[cur][col][16 * half + 8];
        }
#pragma unroll
        for (int m = 0; m < 4; ++m) {   // A frag: lane holds row M, K=16*half+e
            const int row = waveM * 64 + m * 16 + r16;
            FragBF fa;
            fa.h[0] = *(const v8bf*)&sA[cur][row][16 * half];
            fa.h[1] = *(const v8bf*)&sA[cur][row][16 * half + 8];
#pragma unroll
            for (int n = 0; n < 2; ++n)
                acc[m][n] = __builtin_amdgcn_wmma_f32_16x16x32_bf16(
                    false, fa.v, false, fb[n].v, (short)0, acc[m][n], false, false);
        }

        // Stage next panel into the other buffer (not read by anyone now);
        // single barrier makes it visible for iteration kt+1.
        if (kt + 1 < nk) {
            uint4* da = (uint4*)&sA[cur ^ 1][ldRow][ldSeg];
            da[0] = ra0; da[1] = ra1;
            uint4* db = (uint4*)&sB[cur ^ 1][ldRow][ldSeg];
            db[0] = rb0; db[1] = rb1;
            __syncthreads();
        }
    }

    // Epilogue: C layout -> VGPR i holds row (i + 8*half), col = lane&15
#pragma unroll
    for (int m = 0; m < 4; ++m)
#pragma unroll
        for (int n = 0; n < 2; ++n) {
            AccF a; a.v = acc[m][n];
            const int col = n0 + waveN * 32 + n * 16 + r16;
            const float bv = BIAS ? bias[col] : 0.0f;
#pragma unroll
            for (int i = 0; i < 8; ++i) {
                const int row = m0 + waveM * 64 + m * 16 + i + 8 * half;
                float v = a.f[i] + bv;
                if (LEAKY) v = (v >= 0.0f) ? v : 0.01f * v;
                if (OUT_BF16)
                    ((__bf16*)Cout)[(size_t)row * N + col] = (__bf16)v;
                else
                    ((float*)Cout)[(size_t)row * N + col] = v;
            }
        }
}

// ---------------------------------------------------------------------------
// Flash-style causal attention, one block per 16 query rows.
// Full 16x2048 f32 score row lives in (dynamic) LDS: 320KB WGP LDS pays off.
// Phase 1: S = Q·K^T. Subtiles processed in PAIRS with independent fragment
//          registers: 4 global_load_b128 issue as one clause, one wait covers
//          two back-to-back WMMAs (halves stall points vs load/wait/wmma).
// Phase 2: O = P·V, same pairing (P converted bf16 on fragment build;
//          V pre-transposed so B-fragments are contiguous 32B runs).
// ---------------------------------------------------------------------------
__global__ void __launch_bounds__(256)
attn_kernel(const __bf16* __restrict__ q,  // [B][T][H]
            const __bf16* __restrict__ k,  // [B][T][H]
            const __bf16* __restrict__ vT, // [B][H][T]
            __bf16* __restrict__ o) {      // [B][T][H]
    extern __shared__ char smem[];
    const int SS    = Tt + 8;                               // f32 row stride
    float* sS       = (float*)smem;                         // [16][SS]
    float* red      = (float*)(smem + (size_t)16 * SS * 4); // [16][16]
    float* rowsum   = red + 16 * 16;                        // [16]
    float* rowmax   = rowsum + 16;                          // [16]

    const int t    = threadIdx.x;
    const int lane = t & 31;
    const int wid  = t >> 5;       // 0..7
    const int half = lane >> 4;
    const int r16  = lane & 15;

    const int blk   = blockIdx.x;          // 0..255
    const int batch = blk / (Tt / 16);
    const int qt    = blk % (Tt / 16);
    const int qbase = qt * 16;

    const __bf16* qb = q  + (size_t)batch * Tt * Hh;
    const __bf16* kb = k  + (size_t)batch * Tt * Hh;
    const __bf16* vb = vT + (size_t)batch * Hh * Tt;
    __bf16*       ob = o  + (size_t)batch * Tt * Hh;

    const v8f vz = {};

    // ---- Phase 1: scores. Each wave owns 256 cols = 16 N-subtiles. ----
    v8f acc[16];
#pragma unroll
    for (int i = 0; i < 16; ++i) acc[i] = vz;
    const int nwbase = wid * 256;

    for (int h0 = 0; h0 < Hh; h0 += 32) {
        FragBF fa;                      // Q rows (shared across 16 subtiles)
        const __bf16* qrow = qb + (size_t)(qbase + r16) * Hh + h0;
        fa.h[0] = *(const v8bf*)(qrow + 16 * half);
        fa.h[1] = *(const v8bf*)(qrow + 16 * half + 8);

        const __bf16* kbase = kb + (size_t)(nwbase + r16) * Hh + h0 + 16 * half;
#pragma unroll
        for (int np = 0; np < 8; ++np) {     // pairs of N-subtiles
            const int nst = np * 2;
            const __bf16* k0r = kbase + (size_t)nst * 16 * Hh;
            const __bf16* k1r = k0r + (size_t)16 * Hh;
            FragBF f0, f1;                   // independent regs -> one clause
            f0.h[0] = *(const v8bf*)(k0r);
            f0.h[1] = *(const v8bf*)(k0r + 8);
            f1.h[0] = *(const v8bf*)(k1r);
            f1.h[1] = *(const v8bf*)(k1r + 8);
            acc[nst] = __builtin_amdgcn_wmma_f32_16x16x32_bf16(
                false, fa.v, false, f0.v, (short)0, acc[nst], false, false);
            acc[nst + 1] = __builtin_amdgcn_wmma_f32_16x16x32_bf16(
                false, fa.v, false, f1.v, (short)0, acc[nst + 1], false, false);
        }
    }
    const float scale = 0.03125f; // H^-0.5 = 1/32
#pragma unroll
    for (int nst = 0; nst < 16; ++nst) {
        AccF a; a.v = acc[nst];
        const int col = nwbase + nst * 16 + r16;
#pragma unroll
        for (int i = 0; i < 8; ++i) {
            const int m = i + 8 * half;
            float v = a.f[i] * scale;
            if (col > qbase + m) v = -3.0e38f;   // causal mask
            sS[m * SS + col] = v;
        }
    }
    __syncthreads();

    // ---- Softmax over 2048 cols (store unnormalized exp, keep row sums) ----
    {
        const int m   = t >> 4;             // 0..15 (row)
        const int sub = t & 15;             // 0..15 (128-col slice)
        const int cb  = sub * (Tt / 16);
        float mx = -3.4e38f;
        for (int c = 0; c < Tt / 16; ++c) mx = fmaxf(mx, sS[m * SS + cb + c]);
        red[m * 16 + sub] = mx;
        __syncthreads();
        if (sub == 0) {
            float v = red[m * 16];
            for (int j = 1; j < 16; ++j) v = fmaxf(v, red[m * 16 + j]);
            rowmax[m] = v;
        }
        __syncthreads();
        const float rm = rowmax[m];
        float sum = 0.0f;
        for (int c = 0; c < Tt / 16; ++c) {
            const float e = __expf(sS[m * SS + cb + c] - rm);
            sS[m * SS + cb + c] = e;
            sum += e;
        }
        red[m * 16 + sub] = sum;
        __syncthreads();
        if (sub == 0) {
            float v = 0.0f;
            for (int j = 0; j < 16; ++j) v += red[m * 16 + j];
            rowsum[m] = v;
        }
        __syncthreads();
    }

    // ---- Phase 2: O = P·V. Each wave owns 128 head dims = 8 subtiles. ----
    v8f acc2[8];
#pragma unroll
    for (int i = 0; i < 8; ++i) acc2[i] = vz;
    const int dwbase = wid * 128;

    for (int n0 = 0; n0 < Tt; n0 += 32) {
        FragBF fa;                        // P rows, f32->bf16 on the fly
#pragma unroll
        for (int e = 0; e < 16; ++e) {
            const int kk = n0 + e + 16 * half;
            fa.e[e] = (__bf16)sS[r16 * SS + kk];
        }
        const __bf16* vbase = vb + (size_t)(dwbase + r16) * Tt + n0 + 16 * half;
#pragma unroll
        for (int sp = 0; sp < 4; ++sp) {     // pairs of D-subtiles
            const int st = sp * 2;
            const __bf16* v0r = vbase + (size_t)st * 16 * Tt;
            const __bf16* v1r = v0r + (size_t)16 * Tt;
            FragBF f0, f1;
            f0.h[0] = *(const v8bf*)(v0r);
            f0.h[1] = *(const v8bf*)(v0r + 8);
            f1.h[0] = *(const v8bf*)(v1r);
            f1.h[1] = *(const v8bf*)(v1r + 8);
            acc2[st] = __builtin_amdgcn_wmma_f32_16x16x32_bf16(
                false, fa.v, false, f0.v, (short)0, acc2[st], false, false);
            acc2[st + 1] = __builtin_amdgcn_wmma_f32_16x16x32_bf16(
                false, fa.v, false, f1.v, (short)0, acc2[st + 1], false, false);
        }
    }
#pragma unroll
    for (int st = 0; st < 8; ++st) {
        AccF a; a.v = acc2[st];
        const int d = dwbase + st * 16 + r16;
#pragma unroll
        for (int i = 0; i < 8; ++i) {
            const int m = i + 8 * half;
            ob[(size_t)(qbase + m) * Hh + d] = (__bf16)(a.f[i] / rowsum[m]);
        }
    }
}

// ---------------------------------------------------------------------------
// Cross-entropy: loss = mean(lse(row) - row[target]) over 4096 rows
// ---------------------------------------------------------------------------
__global__ void zero_loss(float* loss) { *loss = 0.0f; }

__global__ void __launch_bounds__(256)
ce_loss_kernel(const float* __restrict__ logits, const int* __restrict__ tgt,
               float* __restrict__ loss) {
    __shared__ float red[256];
    const int r = blockIdx.x;
    const float* row = logits + (size_t)r * Vv;
    float mx = -3.4e38f;
    for (int c = threadIdx.x; c < Vv; c += 256) mx = fmaxf(mx, row[c]);
    red[threadIdx.x] = mx;
    __syncthreads();
    for (int s = 128; s > 0; s >>= 1) {
        if (threadIdx.x < s)
            red[threadIdx.x] = fmaxf(red[threadIdx.x], red[threadIdx.x + s]);
        __syncthreads();
    }
    const float rmx = red[0];
    __syncthreads();
    float sum = 0.0f;
    for (int c = threadIdx.x; c < Vv; c += 256) sum += __expf(row[c] - rmx);
    red[threadIdx.x] = sum;
    __syncthreads();
    for (int s = 128; s > 0; s >>= 1) {
        if (threadIdx.x < s) red[threadIdx.x] += red[threadIdx.x + s];
        __syncthreads();
    }
    if (threadIdx.x == 0) {
        const float lse = rmx + __logf(red[0]);
        const float val = lse - row[tgt[r]];
        atomicAdd(loss, val * (1.0f / (float)Mm));
    }
}

// ---------------------------------------------------------------------------
// Host orchestration (graph-capture safe: only kernel launches on `stream`)
// ---------------------------------------------------------------------------
extern "C" void kernel_launch(void* const* d_in, const int* in_sizes, int n_in,
                              void* d_out, int out_size, void* d_ws, size_t ws_size,
                              hipStream_t stream) {
    (void)in_sizes; (void)n_in; (void)out_size; (void)ws_size;
    const int*   idx = (const int*)d_in[0];
    const int*   tgt = (const int*)d_in[1];
    const float* tok = (const float*)d_in[2];
    const float* pos = (const float*)d_in[3];
    const float* wq  = (const float*)d_in[4];
    const float* wk  = (const float*)d_in[5];
    const float* wv  = (const float*)d_in[6];
    const float* hw  = (const float*)d_in[7];   // [L][H][H]
    const float* hb  = (const float*)d_in[8];   // [L][H]
    const float* lw  = (const float*)d_in[9];   // [H][V]
    const float* lb  = (const float*)d_in[10];  // [V]

    float* logits = (float*)d_out;                       // [4096][32000]
    float* loss   = logits + (size_t)Mm * Vv;            // +1 scalar

    // Workspace carve-up (~148 MB total)
    char* wp = (char*)d_ws;
    auto alloc = [&](size_t bytes) -> char* {
        char* p = wp;
        wp += (bytes + 255) & ~(size_t)255;
        return p;
    };
    const size_t actB = (size_t)Mm * Hh * sizeof(__bf16);     // 8 MB each
    __bf16* x_bf  = (__bf16*)alloc(actB);
    __bf16* q_bf  = (__bf16*)alloc(actB);
    __bf16* k_bf  = (__bf16*)alloc(actB);
    __bf16* v_bf  = (__bf16*)alloc(actB);
    __bf16* vT_bf = (__bf16*)alloc(actB);
    __bf16* a_bf  = (__bf16*)alloc(actB);
    __bf16* m0_bf = (__bf16*)alloc(actB);
    __bf16* m1_bf = (__bf16*)alloc(actB);
    __bf16* wqT   = (__bf16*)alloc((size_t)Hh * Hh * 2);
    __bf16* wkT   = (__bf16*)alloc((size_t)Hh * Hh * 2);
    __bf16* wvT   = (__bf16*)alloc((size_t)Hh * Hh * 2);
    __bf16* hwT   = (__bf16*)alloc((size_t)Ll * Hh * Hh * 2);
    __bf16* lwT   = (__bf16*)alloc((size_t)Vv * Hh * 2);      // 65.5 MB

    zero_loss<<<dim3(1), dim3(1), 0, stream>>>(loss);

    // Embedding -> bf16 activations
    embed_kernel<<<dim3(Mm), dim3(256), 0, stream>>>(idx, tok, pos, x_bf);

    // Weight prep: [K][N] f32 -> [N][K] bf16 (K-contiguous B fragments)
    transpose_f32_to_bf16<<<dim3(Hh / 32, Hh / 32), 256, 0, stream>>>(wq, wqT, Hh, Hh);
    transpose_f32_to_bf16<<<dim3(Hh / 32, Hh / 32), 256, 0, stream>>>(wk, wkT, Hh, Hh);
    transpose_f32_to_bf16<<<dim3(Hh / 32, Hh / 32), 256, 0, stream>>>(wv, wvT, Hh, Hh);
    for (int i = 0; i < Ll; ++i)
        transpose_f32_to_bf16<<<dim3(Hh / 32, Hh / 32), 256, 0, stream>>>(
            hw + (size_t)i * Hh * Hh, hwT + (size_t)i * Hh * Hh, Hh, Hh);
    transpose_f32_to_bf16<<<dim3(Vv / 32, Hh / 32), 256, 0, stream>>>(lw, lwT, Hh, Vv);

    // QKV projections (bf16 out, no bias/act)
    const dim3 gHH(Hh / 128, Mm / 128);
    gemm_bf16_wmma<Hh, true, false, false><<<gHH, 256, 0, stream>>>(x_bf, wqT, nullptr, q_bf, Mm, Hh);
    gemm_bf16_wmma<Hh, true, false, false><<<gHH, 256, 0, stream>>>(x_bf, wkT, nullptr, k_bf, Mm, Hh);
    gemm_bf16_wmma<Hh, true, false, false><<<gHH, 256, 0, stream>>>(x_bf, wvT, nullptr, v_bf, Mm, Hh);

    // V -> V^T per batch ([T][H] -> [H][T]) for contiguous PV B-fragments
    transpose_bf16<<<dim3(Hh / 32, Tt / 32, Bb), 256, 0, stream>>>(v_bf, vT_bf, Tt, Hh);

    // Causal attention (dynamic LDS: 16x(T+8) f32 scores + reductions)
    const size_t attnLds = ((size_t)16 * (Tt + 8) + 16 * 16 + 32) * sizeof(float);
    attn_kernel<<<dim3(Bb * Tt / 16), 256, attnLds, stream>>>(q_bf, k_bf, vT_bf, a_bf);

    // MLP stack: Linear + bias + LeakyReLU, bf16 ping-pong
    const __bf16* cur = a_bf;
    __bf16* bufs[2] = {m0_bf, m1_bf};
    for (int i = 0; i < Ll; ++i) {
        __bf16* outb = bufs[i & 1];
        gemm_bf16_wmma<Hh, true, true, true><<<gHH, 256, 0, stream>>>(
            cur, hwT + (size_t)i * Hh * Hh, hb + (size_t)i * Hh, outb, Mm, Hh);
        cur = outb;
    }

    // Logits: f32 out straight to d_out, + bias
    gemm_bf16_wmma<Hh, false, true, false><<<dim3(Vv / 128, Mm / 128), 256, 0, stream>>>(
        cur, lwT, lb, (void*)logits, Mm, Vv);

    // Cross-entropy mean
    ce_loss_kernel<<<dim3(Mm), 256, 0, stream>>>(logits, tgt, loss);
}
